// RGCN_Layer_50603304681880
// MI455X (gfx1250) — matrix operand order repaired and checked by reading, not compiled, not run on hardware
//
#include <hip/hip_runtime.h>

typedef __attribute__((ext_vector_type(8)))  float         v8f;
typedef __attribute__((ext_vector_type(16))) __bf16        v16bf;
typedef __attribute__((ext_vector_type(4)))  float         f32x4;
typedef __attribute__((ext_vector_type(4)))  unsigned int  u32x4;

#define N_    4096
#define D_    128
#define R_    8
#define OUT_  128
#define B_    4096
#define KTOT  (D_ + R_ * N_)   // 32896
#define BM    32               // rows per block in main kernel

__device__ __forceinline__ unsigned short f32_to_bf16(float f) {
    unsigned int u = __float_as_uint(f);
    unsigned int r = u + 0x7FFFu + ((u >> 16) & 1u);  // round-to-nearest-even
    return (unsigned short)(r >> 16);
}

union BV { u32x4 q[2]; v16bf v; };

// ---------------------------------------------------------------------------
// Phase 1: build Wcat (K=32896 x OUT=128) in bf16, pre-swizzled into a
// CDNA5 WMMA B-operand (32x16, 16-bit) lane layout, in this file's contiguous k order:
//   tile (kt, ot): lane l, element e  <->  K = kt*32 + 16*(l>=16) + e,
//                                          O = ot*16 + (l & 15)
//   stored at  ws[(kt*8 + ot)*512 + l*16 + e]
// Rows K<128 hold self_kernel; K=128+r*4096+n holds W_r[n,o]=sum_d emb[n,d]*rk[r,d,o].
// ---------------------------------------------------------------------------
__global__ void __launch_bounds__(256)
rgcn_build_w(const float* __restrict__ emb,
             const float* __restrict__ rk,
             const float* __restrict__ selfk,
             unsigned short* __restrict__ wswz)
{
    size_t t   = (size_t)blockIdx.x * 256 + threadIdx.x;   // == swizzled offset
    int   e_   = (int)(t & 15);
    int   l    = (int)((t >> 4) & 31);
    size_t til = t >> 9;
    size_t kt  = til >> 3;
    int   ot   = (int)(til & 7);
    int   k    = (int)(kt * 32) + ((l >= 16) ? 16 : 0) + e_;
    int   o    = ot * 16 + (l & 15);

    float val;
    if (k < D_) {
        val = selfk[(size_t)k * OUT_ + o];
    } else {
        int k2 = k - D_;
        int r  = k2 >> 12;          // /4096
        int n  = k2 & (N_ - 1);
        const float* er  = emb + (size_t)n * D_;
        const float* rkc = rk + (size_t)r * D_ * OUT_ + o;
        float s = 0.f;
        #pragma unroll 4
        for (int d = 0; d < D_; ++d)
            s += er[d] * rkc[(size_t)d * OUT_];
        val = s;
    }
    wswz[t] = f32_to_bf16(val);
}

// ---------------------------------------------------------------------------
// Phase 2: C(4096x128) = [e | gathered adj rows] (bf16) x Wcat (bf16), fp32 acc.
// grid = (B/BM, 2 branches), block = 256 (8 waves).
// Wave w -> O-tile w; block covers 32 rows = 2 M-tiles per wave.
// ---------------------------------------------------------------------------
__global__ void __launch_bounds__(256)
rgcn_main(const float* __restrict__ adj,
          const int*   __restrict__ head_idx,
          const float* __restrict__ head_e,
          const int*   __restrict__ tail_idx,
          const float* __restrict__ tail_e,
          const unsigned short* __restrict__ wswz,
          float* __restrict__ out)
{
    __shared__ unsigned short lds_a[BM][64];   // 32 rows x 64 K of bf16 (4 KB)
    __shared__ int sidx[BM];

    const int  branch = blockIdx.y;
    const int* idx    = branch ? tail_idx : head_idx;
    const float* ee   = branch ? tail_e   : head_e;
    float* outp       = out + (size_t)branch * B_ * OUT_;

    const int tid  = threadIdx.x;
    const int lane = tid & 31;
    const int wave = tid >> 5;               // 0..7 -> O tile
    const int m0   = blockIdx.x * BM;

    if (tid < BM) sidx[tid] = idx[m0 + tid];
    __syncthreads();

    v8f acc0 = {};   // rows m0 .. m0+15
    v8f acc1 = {};   // rows m0+16 .. m0+31

    // fill mapping: 8 threads per row, 8 floats (two f32x4) each
    const int row  = tid >> 3;               // 0..31
    const int kk   = (tid & 7) * 8;          // 0..56
    const int gRow = m0 + row;

    const int mrow = lane & 15;
    const int koff = (lane < 16) ? 0 : 16;

    for (int kb = 0; kb < KTOT / 64; ++kb) {
        // ---- stage A block (32 rows x 64 K) fp32 -> bf16 -> LDS ----
        int k = kb * 64 + kk;                // 64-chunks never straddle regions
        const float* src;
        if (k < D_) {
            src = ee + (size_t)gRow * D_ + k;
        } else {
            int k2 = k - D_;
            int r  = k2 >> 12;
            int n  = k2 & (N_ - 1);
            src = adj + (((size_t)r * N_ + (unsigned)sidx[row]) * N_ + n);
        }
        f32x4 f0 = __builtin_nontemporal_load((const f32x4*)src);
        f32x4 f1 = __builtin_nontemporal_load((const f32x4*)src + 1);

        // prefetch next K block for this thread's row
        int kn = (kb + 1) * 64 + kk;
        if (kn < KTOT) {
            const float* psrc;
            if (kn < D_) {
                psrc = ee + (size_t)gRow * D_ + kn;
            } else {
                int k2 = kn - D_;
                int r  = k2 >> 12;
                int n  = k2 & (N_ - 1);
                psrc = adj + (((size_t)r * N_ + (unsigned)sidx[row]) * N_ + n);
            }
            __builtin_prefetch(psrc, 0, 1);
        }

        unsigned short* dst = &lds_a[row][kk];
        dst[0] = f32_to_bf16(f0.x); dst[1] = f32_to_bf16(f0.y);
        dst[2] = f32_to_bf16(f0.z); dst[3] = f32_to_bf16(f0.w);
        dst[4] = f32_to_bf16(f1.x); dst[5] = f32_to_bf16(f1.y);
        dst[6] = f32_to_bf16(f1.z); dst[7] = f32_to_bf16(f1.w);
        __syncthreads();

        // ---- two 16x16x32 WMMA sub-steps ----
        #pragma unroll
        for (int s = 0; s < 2; ++s) {
            int kt = kb * 2 + s;

            // B operand: contiguous 32B per lane from pre-swizzled Wcat (L2 hot)
            BV bv;
            const u32x4* bp =
                (const u32x4*)(wswz + ((size_t)kt * 8 + wave) * 512 + lane * 16);
            bv.q[0] = bp[0];
            bv.q[1] = bp[1];

            // A operands from LDS: contiguous k = 16*(lane/16) + e order, same as B
            BV a0, a1;
            a0.q[0] = *(const u32x4*)&lds_a[mrow][s * 32 + koff];
            a0.q[1] = *(const u32x4*)&lds_a[mrow][s * 32 + koff + 8];
            a1.q[0] = *(const u32x4*)&lds_a[16 + mrow][s * 32 + koff];
            a1.q[1] = *(const u32x4*)&lds_a[16 + mrow][s * 32 + koff + 8];

            acc0 = __builtin_amdgcn_wmma_f32_16x16x32_bf16(
                       false, a0.v, false, bv.v, (short)0, acc0, false, false);
            acc1 = __builtin_amdgcn_wmma_f32_16x16x32_bf16(
                       false, a1.v, false, bv.v, (short)0, acc1, false, false);
        }
        __syncthreads();
    }

    // ---- write C: VGPR i, lanes 0-15 -> M=i, lanes 16-31 -> M=8+i; N=lane&15
    const int nC   = lane & 15;
    const int mAdd = (lane >> 4) * 8;
    float* ob = outp + (size_t)(m0 + mAdd) * OUT_ + wave * 16 + nC;
    #pragma unroll
    for (int i = 0; i < 8; ++i) {
        ob[(size_t)i * OUT_]        = acc0[i];
        ob[(size_t)(16 + i) * OUT_] = acc1[i];
    }
}

extern "C" void kernel_launch(void* const* d_in, const int* in_sizes, int n_in,
                              void* d_out, int out_size, void* d_ws, size_t ws_size,
                              hipStream_t stream) {
    const float* emb      = (const float*)d_in[0];
    const int*   head_idx = (const int*)  d_in[1];
    const float* head_e   = (const float*)d_in[2];
    const int*   tail_idx = (const int*)  d_in[3];
    const float* tail_e   = (const float*)d_in[4];
    const float* adj      = (const float*)d_in[5];
    const float* rk       = (const float*)d_in[6];
    const float* selfk    = (const float*)d_in[7];
    float* out            = (float*)d_out;
    unsigned short* wswz  = (unsigned short*)d_ws;

    const size_t ws_needed = (size_t)KTOT * OUT_ * sizeof(unsigned short); // ~8.4 MB
    if (ws_size < ws_needed) return;

    // Phase 1: 1028 K-tiles * 8 O-tiles * 512 elems = 4,210,688 threads
    const int p1_blocks = (KTOT / 32) * 8 * 512 / 256;   // 16448
    rgcn_build_w<<<p1_blocks, 256, 0, stream>>>(emb, rk, selfk, wswz);

    // Phase 2: 128 row-blocks x 2 branches
    dim3 grid(B_ / BM, 2);
    rgcn_main<<<grid, 256, 0, stream>>>(adj, head_idx, head_e,
                                        tail_idx, tail_e, wswz, out);
}
